// ActorCriticPolicy_50981261803603
// MI455X (gfx1250) — matrix operand order repaired and checked by reading, not compiled, not run on hardware
//
#include <hip/hip_runtime.h>

#define DEV static __device__ __forceinline__

typedef __bf16 v16bf __attribute__((ext_vector_type(16)));
typedef __bf16 v2bf  __attribute__((ext_vector_type(2)));
typedef float  v8f   __attribute__((ext_vector_type(8)));
typedef int    v4i   __attribute__((ext_vector_type(4)));

struct U8 { unsigned d[8]; };   // dword view of a 32-byte fragment
struct Q2 { uint4 a, b; };      // two-uint4 view of a 32-byte fragment

#if defined(__has_builtin)
#if __has_builtin(__builtin_amdgcn_global_load_async_to_lds_b128) && \
    __has_builtin(__builtin_amdgcn_s_wait_asynccnt)
#define USE_ASYNC_LDS 1
#endif
#endif

#define AS1 __attribute__((address_space(1)))
#define AS3 __attribute__((address_space(3)))

// ---------- bf16 helpers ----------
DEV unsigned short f2bf(float f) {       // host-safe scalar version (prep kernel)
  unsigned u = __float_as_uint(f);
  unsigned r = 0x7FFFu + ((u >> 16) & 1u);
  return (unsigned short)((u + r) >> 16);
}
DEV unsigned pack2(float a, float b) {   // native packed convert on gfx1250
  v2bf p;
  p[0] = (__bf16)a;
  p[1] = (__bf16)b;
  return __builtin_bit_cast(unsigned, p);
}
DEV float bfl(unsigned u) { return __uint_as_float(u << 16); }
DEV float bfh(unsigned u) { return __uint_as_float(u & 0xFFFF0000u); }

DEV v8f vzero() {
  v8f z;
#pragma unroll
  for (int i = 0; i < 8; ++i) z[i] = 0.f;
  return z;
}

DEV void load8(const float* base, float out[8]) {
  const float4* p = reinterpret_cast<const float4*>(base);
  float4 a = p[0], b = p[1];
  out[0]=a.x; out[1]=a.y; out[2]=a.z; out[3]=a.w;
  out[4]=b.x; out[5]=b.y; out[6]=b.z; out[7]=b.w;
}

// ---------- cross-lane pack: acc layout (f = mt*16 + 8*hw + j) -> B fragment
// B frag k: lane holds K-window features 32k + 16*hw + [0..15], 2 bf16/dword.
DEV v16bf pack_frag(const v8f& t0, const v8f& t1, int hw) {
  unsigned od[4], sd[4], rd[4];
#pragma unroll
  for (int j = 0; j < 4; ++j) {
    float o0 = hw ? t1[2*j]   : t0[2*j];
    float o1 = hw ? t1[2*j+1] : t0[2*j+1];
    float s0 = hw ? t0[2*j]   : t1[2*j];
    float s1 = hw ? t0[2*j+1] : t1[2*j+1];
    od[j] = pack2(o0, o1);
    sd[j] = pack2(s0, s1);
  }
#pragma unroll
  for (int j = 0; j < 4; ++j) rd[j] = (unsigned)__shfl_xor((int)sd[j], 16, 32);
  U8 f;
#pragma unroll
  for (int j = 0; j < 4; ++j) {
    f.d[j]     = hw ? rd[j] : od[j];
    f.d[4 + j] = hw ? od[j] : rd[j];
  }
  return __builtin_bit_cast(v16bf, f);
}

DEV void unpack_frag(const v16bf& fv, int hw, v8f& h0, v8f& h1) {
  U8 f = __builtin_bit_cast(U8, fv);
  unsigned sd[4], rd[4];
#pragma unroll
  for (int j = 0; j < 4; ++j) sd[j] = hw ? f.d[j] : f.d[4 + j];
#pragma unroll
  for (int j = 0; j < 4; ++j) rd[j] = (unsigned)__shfl_xor((int)sd[j], 16, 32);
#pragma unroll
  for (int j = 0; j < 4; ++j) {
    unsigned uo = hw ? f.d[4 + j] : f.d[j];
    float oa = bfl(uo), ob = bfh(uo);
    float ra = bfl(rd[j]), rb = bfh(rd[j]);
    h0[2*j]   = hw ? ra : oa;
    h0[2*j+1] = hw ? rb : ob;
    h1[2*j]   = hw ? oa : ra;
    h1[2*j+1] = hw ? ob : rb;
  }
}

// ---------- GEMM: acc[mt](16 feat x 16 samples) += WT_tile x Bfrag over NK ksteps
// wT4: bf16 [NMT*16][NK*32] as uint4 rows; LDS double-buffered 16KB strips.
template<int NK, int NMT>
DEV void gemm_layer(const uint4* __restrict__ wT4, int rowStride4,
                    const v16bf* bfrag, v8f* acc, uint4* lds,
                    int tid, int lane) {
  const int nrows = NMT * 16;
  const int hw = (lane >> 4) & 1;
  const int mrow = lane & 15;
#pragma unroll
  for (int mt = 0; mt < NMT; ++mt) acc[mt] = vzero();

  const bool act = tid < nrows;

#if USE_ASYNC_LDS
  // Async global->LDS staging (ASYNCcnt); 4x b128 per thread per strip.
#pragma unroll
  for (int q = 0; q < 4; ++q)
    if (act)
      __builtin_amdgcn_global_load_async_to_lds_b128(
          (AS1 v4i*)(unsigned long long)(wT4 + (size_t)tid * rowStride4 + q),
          (AS3 v4i*)(unsigned long long)(unsigned)(unsigned long long)(void*)(lds + tid * 4 + q),
          0, 0);
#pragma unroll
  for (int k = 0; k < NK; ++k) {
    if (k + 1 < NK) {
#pragma unroll
      for (int q = 0; q < 4; ++q)
        if (act)
          __builtin_amdgcn_global_load_async_to_lds_b128(
              (AS1 v4i*)(unsigned long long)(wT4 + (size_t)tid * rowStride4 + (k + 1) * 4 + q),
              (AS3 v4i*)(unsigned long long)(unsigned)(unsigned long long)(void*)(lds + ((k + 1) & 1) * 1024 + tid * 4 + q),
              0, 0);
      __builtin_amdgcn_s_wait_asynccnt(4);
    } else {
      __builtin_amdgcn_s_wait_asynccnt(0);
    }
    __syncthreads();
    const uint4* buf = lds + (k & 1) * 1024;
    // software-pipelined A-fragment loads (1 deep)
    Q2 q;
    q.a = buf[mrow * 4 + 2 * hw];        // K = 16*hw .. 16*hw+7
    q.b = buf[mrow * 4 + 2 * hw + 1];    // K = 16*hw+8 .. 16*hw+15
#pragma unroll
    for (int mt = 0; mt < NMT; ++mt) {
      Q2 qn = q;
      if (mt + 1 < NMT) {
        int row = (mt + 1) * 16 + mrow;
        qn.a = buf[row * 4 + 2 * hw];
        qn.b = buf[row * 4 + 2 * hw + 1];
      }
      acc[mt] = __builtin_amdgcn_wmma_f32_16x16x32_bf16(
          false, __builtin_bit_cast(v16bf, q), false, bfrag[k], (short)0, acc[mt],
          false, false);
      q = qn;
    }
    __syncthreads();
  }
#else
  uint4 r0, r1, r2, r3;
  if (act) {
    const uint4* s = wT4 + (size_t)tid * rowStride4;
    r0 = s[0]; r1 = s[1]; r2 = s[2]; r3 = s[3];
  }
#pragma unroll
  for (int k = 0; k < NK; ++k) {
    uint4* buf = lds + (k & 1) * 1024;
    if (act) {
      uint4* d = buf + tid * 4;
      d[0] = r0; d[1] = r1; d[2] = r2; d[3] = r3;
    }
    __syncthreads();
    if (k + 1 < NK && act) {
      const uint4* s = wT4 + (size_t)tid * rowStride4 + (k + 1) * 4;
      r0 = s[0]; r1 = s[1]; r2 = s[2]; r3 = s[3];
    }
    if (k + 2 < NK && act)
      __builtin_prefetch(wT4 + (size_t)tid * rowStride4 + (k + 2) * 4, 0, 1);
    Q2 q;
    q.a = buf[mrow * 4 + 2 * hw];
    q.b = buf[mrow * 4 + 2 * hw + 1];
#pragma unroll
    for (int mt = 0; mt < NMT; ++mt) {
      Q2 qn = q;
      if (mt + 1 < NMT) {
        int row = (mt + 1) * 16 + mrow;
        qn.a = buf[row * 4 + 2 * hw];
        qn.b = buf[row * 4 + 2 * hw + 1];
      }
      acc[mt] = __builtin_amdgcn_wmma_f32_16x16x32_bf16(
          false, __builtin_bit_cast(v16bf, q), false, bfrag[k], (short)0, acc[mt],
          false, false);
      q = qn;
    }
    __syncthreads();
  }
#endif
}

// ---------- epilogue pieces ----------
template<int NMT>
DEV void bias_relu_stats(v8f* acc, const float* __restrict__ bias, int hw,
                         float& mean, float& rstd) {
  float sum = 0.f, sq = 0.f;
#pragma unroll
  for (int mt = 0; mt < NMT; ++mt) {
    float bb[8]; load8(bias + mt * 16 + hw * 8, bb);
#pragma unroll
    for (int j = 0; j < 8; ++j) {
      float v = fmaxf(acc[mt][j] + bb[j], 0.f);
      acc[mt][j] = v;
      sum += v; sq += v * v;
    }
  }
  sum += __shfl_xor(sum, 16, 32);
  sq  += __shfl_xor(sq , 16, 32);
  const float inv_n = 1.f / (float)(NMT * 16);
  mean = sum * inv_n;
  float var = sq * inv_n - mean * mean;
  rstd = rsqrtf(var + 1e-5f);
}

template<int NMT, bool RES>
DEV void ln_pack(v8f* acc, const float* __restrict__ g, const float* __restrict__ be,
                 float mean, float rstd, int hw, v16bf* frag) {
#pragma unroll
  for (int k = 0; k < NMT / 2; ++k) {
    v8f t0, t1;
    float gg[8], ee[8];
    load8(g + (2*k) * 16 + hw * 8, gg); load8(be + (2*k) * 16 + hw * 8, ee);
#pragma unroll
    for (int j = 0; j < 8; ++j) t0[j] = (acc[2*k][j] - mean) * rstd * gg[j] + ee[j];
    load8(g + (2*k+1) * 16 + hw * 8, gg); load8(be + (2*k+1) * 16 + hw * 8, ee);
#pragma unroll
    for (int j = 0; j < 8; ++j) t1[j] = (acc[2*k+1][j] - mean) * rstd * gg[j] + ee[j];
    if (RES) {
      v8f h0, h1;
      unpack_frag(frag[k], hw, h0, h1);
#pragma unroll
      for (int j = 0; j < 8; ++j) { t0[j] += h0[j]; t1[j] += h1[j]; }
    }
    frag[k] = pack_frag(t0, t1, hw);
  }
}

template<int NMT>
DEV void bias_relu_pack(v8f* acc, const float* __restrict__ bias, int hw, v16bf* frag) {
#pragma unroll
  for (int k = 0; k < NMT / 2; ++k) {
    v8f t0, t1;
    float bb[8];
    load8(bias + (2*k) * 16 + hw * 8, bb);
#pragma unroll
    for (int j = 0; j < 8; ++j) t0[j] = fmaxf(acc[2*k][j] + bb[j], 0.f);
    load8(bias + (2*k+1) * 16 + hw * 8, bb);
#pragma unroll
    for (int j = 0; j < 8; ++j) t1[j] = fmaxf(acc[2*k+1][j] + bb[j], 0.f);
    frag[k] = pack_frag(t0, t1, hw);
  }
}

DEV void build_input_frags(const float* __restrict__ xrow, int hw, v16bf* frag) {
#pragma unroll
  for (int k = 0; k < 4; ++k) {
    int f0 = k * 32 + hw * 16;
    U8 u;
#pragma unroll
    for (int j = 0; j < 8; ++j) {
      int f = f0 + 2 * j;
      float a = (f     < 100) ? xrow[f]     : 0.f;
      float b = (f + 1 < 100) ? xrow[f + 1] : 0.f;
      u.d[j] = pack2(a, b);
    }
    frag[k] = __builtin_bit_cast(v16bf, u);
  }
}

// ---------- shared encoder trunk ----------
struct EncParams {
  const uint4* wT_in;    // bf16 [256][128]
  const uint4* wT_blk;   // bf16 4 x [256][256]
  const float *b_in, *g_in, *be_in;
  const float *Bl, *Gl, *Bel;   // [4][256] each
};

DEV void encode(const float* __restrict__ xrow, const EncParams& p,
                v16bf* frag, v8f* acc, uint4* lds, int tid, int lane, int hw) {
  build_input_frags(xrow, hw, frag);
  gemm_layer<4, 16>(p.wT_in, 16, frag, acc, lds, tid, lane);
  float mean, rstd;
  bias_relu_stats<16>(acc, p.b_in, hw, mean, rstd);
  ln_pack<16, false>(acc, p.g_in, p.be_in, mean, rstd, hw, frag);
  for (int l = 0; l < 4; ++l) {
    gemm_layer<8, 16>(p.wT_blk + (size_t)l * 8192, 32, frag, acc, lds, tid, lane);
    bias_relu_stats<16>(acc, p.Bl + l * 256, hw, mean, rstd);
    ln_pack<16, true>(acc, p.Gl + l * 256, p.Bel + l * 256, mean, rstd, hw, frag);
  }
}

// ---------- kernels ----------
__global__ void prep_weight(const float* __restrict__ src, unsigned short* __restrict__ dst,
                            int K, int N, int Kpad, int Npad) {
  int idx = blockIdx.x * blockDim.x + threadIdx.x;
  if (idx >= Kpad * Npad) return;
  int n = idx / Kpad, k = idx % Kpad;
  float v = (k < K && n < N) ? src[(size_t)k * N + n] : 0.f;
  dst[idx] = f2bf(v);
}

// MODE 0: obs -> enc_obs (padded [1024][112]); MODE 1: actions -> masked logits
template<int MODE>
__global__ __launch_bounds__(256, 1)
void policy_kernel(const float* __restrict__ x, EncParams p,
                   const uint4* __restrict__ wT_w1, const uint4* __restrict__ wT_w2,
                   const float* __restrict__ b1, const float* __restrict__ b2,
                   float* __restrict__ enc_obs_w, const float* __restrict__ enc_obs_r,
                   const int* __restrict__ mask, float* __restrict__ logits) {
  __shared__ uint4 lds[2048];
  const int tid = threadIdx.x, lane = tid & 31, wid = tid >> 5;
  const int hw = (lane >> 4) & 1, lcol = lane & 15;
  const long s = (long)blockIdx.x * 128 + wid * 16 + lcol;

  v16bf frag[8];
  v8f acc[16];
  encode(x + s * 100, p, frag, acc, lds, tid, lane, hw);

  gemm_layer<8, 16>(wT_w1, 32, frag, acc, lds, tid, lane);
  bias_relu_pack<16>(acc, b1, hw, frag);
  gemm_layer<8, 7>(wT_w2, 32, frag, acc, lds, tid, lane);

  if (MODE == 0) {
#pragma unroll
    for (int mt = 0; mt < 7; ++mt)
#pragma unroll
      for (int j = 0; j < 8; ++j) {
        int f = mt * 16 + hw * 8 + j;
        enc_obs_w[s * 112 + f] = (f < 100) ? (acc[mt][j] + b2[f]) : 0.f;
      }
  } else {
    const int b = (int)(s >> 8);
    const float* ob = enc_obs_r + (long)b * 112;
    float dot = 0.f;
#pragma unroll
    for (int mt = 0; mt < 7; ++mt)
#pragma unroll
      for (int j = 0; j < 8; ++j) {
        int f = mt * 16 + hw * 8 + j;
        float v = acc[mt][j] + ((f < 100) ? b2[f] : 0.f);
        dot += v * ob[f];
      }
    dot += __shfl_xor(dot, 16, 32);
    if (lane < 16)
      logits[s] = mask[s] ? dot * 0.1f : -__builtin_inff();
  }
}

__global__ __launch_bounds__(256, 1)
void value_kernel(const float* __restrict__ x, EncParams p,
                  const uint4* __restrict__ wT_w1, const float* __restrict__ b1,
                  const float* __restrict__ w2, const float* __restrict__ b2,
                  float* __restrict__ out_value) {
  __shared__ uint4 lds[2048];
  const int tid = threadIdx.x, lane = tid & 31, wid = tid >> 5;
  const int hw = (lane >> 4) & 1, lcol = lane & 15;
  const long s = (long)blockIdx.x * 128 + wid * 16 + lcol;

  v16bf frag[8];
  v8f acc[16];
  encode(x + s * 100, p, frag, acc, lds, tid, lane, hw);

  gemm_layer<8, 8>(wT_w1, 32, frag, acc, lds, tid, lane);
  float dot = 0.f;
#pragma unroll
  for (int mt = 0; mt < 8; ++mt) {
    float bb[8], ww[8];
    load8(b1 + mt * 16 + hw * 8, bb);
    load8(w2 + mt * 16 + hw * 8, ww);
#pragma unroll
    for (int j = 0; j < 8; ++j)
      dot += fmaxf(acc[mt][j] + bb[j], 0.f) * ww[j];
  }
  dot += __shfl_xor(dot, 16, 32);
  if (lane < 16) out_value[s] = dot + b2[0];
}

// ---------- host ----------
extern "C" void kernel_launch(void* const* d_in, const int* in_sizes, int n_in,
                              void* d_out, int out_size, void* d_ws, size_t ws_size,
                              hipStream_t stream) {
  (void)in_sizes; (void)n_in; (void)out_size; (void)ws_size;
  const float* obs    = (const float*)d_in[0];
  const float* actemb = (const float*)d_in[1];
  const int*   mask   = (const int*)d_in[2];
  const float* pw_in  = (const float*)d_in[3];
  const float* pb_in  = (const float*)d_in[4];
  const float* pg_in  = (const float*)d_in[5];
  const float* pbe_in = (const float*)d_in[6];
  const float* pW     = (const float*)d_in[7];
  const float* pB     = (const float*)d_in[8];
  const float* pG     = (const float*)d_in[9];
  const float* pBe    = (const float*)d_in[10];
  const float* po_w1  = (const float*)d_in[11];
  const float* po_b1  = (const float*)d_in[12];
  const float* po_w2  = (const float*)d_in[13];
  const float* po_b2  = (const float*)d_in[14];
  const float* vw_in  = (const float*)d_in[15];
  const float* vb_in  = (const float*)d_in[16];
  const float* vg_in  = (const float*)d_in[17];
  const float* vbe_in = (const float*)d_in[18];
  const float* vW     = (const float*)d_in[19];
  const float* vB     = (const float*)d_in[20];
  const float* vG     = (const float*)d_in[21];
  const float* vBe    = (const float*)d_in[22];
  const float* vo_w1  = (const float*)d_in[23];
  const float* vo_b1  = (const float*)d_in[24];
  const float* vo_w2  = (const float*)d_in[25];
  const float* vo_b2  = (const float*)d_in[26];

  char* ws = (char*)d_ws;
  size_t off = 0;
  auto take = [&](size_t bytes) {
    void* p = ws + off;
    off = (off + bytes + 255) & ~(size_t)255;
    return p;
  };
  unsigned short* wTp_in  = (unsigned short*)take(256 * 128 * 2);
  unsigned short* wTp_blk = (unsigned short*)take(4 * 256 * 256 * 2);
  unsigned short* wTp_w1  = (unsigned short*)take(256 * 256 * 2);
  unsigned short* wTp_w2  = (unsigned short*)take(112 * 256 * 2);
  unsigned short* wTv_in  = (unsigned short*)take(256 * 128 * 2);
  unsigned short* wTv_blk = (unsigned short*)take(4 * 256 * 256 * 2);
  unsigned short* wTv_w1  = (unsigned short*)take(128 * 256 * 2);
  float*          enc_obs = (float*)take(1024 * 112 * 4);

  auto prep = [&](const float* src, unsigned short* dst, int K, int N, int Kpad, int Npad) {
    int total = Kpad * Npad;
    prep_weight<<<(total + 255) / 256, 256, 0, stream>>>(src, dst, K, N, Kpad, Npad);
  };
  prep(pw_in, wTp_in, 100, 256, 128, 256);
  for (int l = 0; l < 4; ++l)
    prep(pW + (size_t)l * 65536, wTp_blk + (size_t)l * 65536, 256, 256, 256, 256);
  prep(po_w1, wTp_w1, 256, 256, 256, 256);
  prep(po_w2, wTp_w2, 256, 100, 256, 112);
  prep(vw_in, wTv_in, 100, 256, 128, 256);
  for (int l = 0; l < 4; ++l)
    prep(vW + (size_t)l * 65536, wTv_blk + (size_t)l * 65536, 256, 256, 256, 256);
  prep(vo_w1, wTv_w1, 256, 128, 256, 128);

  EncParams pp = { (const uint4*)wTp_in, (const uint4*)wTp_blk,
                   pb_in, pg_in, pbe_in, pB, pG, pBe };
  EncParams pv = { (const uint4*)wTv_in, (const uint4*)wTv_blk,
                   vb_in, vg_in, vbe_in, vB, vG, vBe };

  float* logits = (float*)d_out;
  float* value  = logits + 1024 * 256;

  policy_kernel<0><<<8, 256, 0, stream>>>(obs, pp, (const uint4*)wTp_w1,
      (const uint4*)wTp_w2, po_b1, po_b2, enc_obs, nullptr, nullptr, nullptr);
  policy_kernel<1><<<2048, 256, 0, stream>>>(actemb, pp, (const uint4*)wTp_w1,
      (const uint4*)wTp_w2, po_b1, po_b2, nullptr, enc_obs, mask, logits);
  value_kernel<<<8, 256, 0, stream>>>(obs, pv, (const uint4*)wTv_w1,
      vo_b1, vo_w2, vo_b2, value);
}
